// STFT_Process_33870112096580
// MI455X (gfx1250) — matrix operand order repaired and checked by reading, not compiled, not run on hardware
//
#include <hip/hip_runtime.h>

typedef __attribute__((ext_vector_type(16))) _Float16 v16h;
typedef __attribute__((ext_vector_type(8)))  float    v8f;

#define N_FFT     400
#define HOP       160
#define HALF_NFFT 200
#define N_FREQ    201
#define N_SAMP    480000
#define N_FRAMES  3001
#define N_BATCH   32

#define KPAD      416                      // K padded to 13 chunks of 32
#define K_CHUNKS  13
#define M_TILES   13                       // 208 rows (201 valid)
#define N_TASKS   26                       // 13 m-tiles x {cos, sin}
#define TASKS_PAD 28                       // padded so every wave owns exactly 7
#define A_ELEMS   (TASKS_PAD * K_CHUNKS * 32 * 16)   // 186368 halves per plane

#define WG_FRAMES 32
#define N32_TILES ((N_FRAMES + WG_FRAMES - 1) / WG_FRAMES)   // 94

// ---------------------------------------------------------------------------
// Prologue: split f32 cos/sin kernels into f16 hi/lo planes, laid out in the
// contiguous per-lane k order this code uses for the WMMA A fragment (B matches):
//   lanes 0-15  hold row M=lane,    j=0..15 -> K=j
//   lanes 16-31 hold row M=lane-16, j=0..15 -> K=j+16
// so the main kernel's A loads are contiguous 32B per lane. Tasks 26/27 are
// zero padding (branch-free task loop in the main kernel).
// ---------------------------------------------------------------------------
__global__ __launch_bounds__(256) void stft_pack_kernels(
    const float* __restrict__ ck, const float* __restrict__ sk,
    _Float16* __restrict__ ahg, _Float16* __restrict__ alg)
{
    int g = blockIdx.x * 256 + threadIdx.x;
    if (g >= A_ELEMS) return;
    int j    = g & 15;
    int rest = g >> 4;
    int lane = rest & 31;
    rest >>= 5;
    int c    = rest % K_CHUNKS;
    int task = rest / K_CHUNKS;            // 0..27 : task = mat*13 + mtile

    int koff = (lane >> 4) * 16 + j;
    int K = c * 32 + koff;
    int M = (task % M_TILES) * 16 + (lane & 15);

    float v = 0.0f;
    if (task < N_TASKS && M < N_FREQ && K < N_FFT) {
        const float* kern = (task / M_TILES) ? sk : ck;
        v = kern[M * N_FFT + K];
    }
    _Float16 h = (_Float16)v;
    _Float16 l = (_Float16)(v - (float)h);
    ahg[g] = h;
    alg[g] = l;
}

// ---------------------------------------------------------------------------
// Main: one workgroup = (batch b, 32 frames). Stage overlapping frame windows
// (with zero padding) into LDS as split-f16 B fragments; 8 waves compute all
// 26 (m-tile x cos/sin) 16x16 output tiles with 3 WMMAs per K-chunk
// (hi*hi + hi*lo + lo*hi), f32 accumulate. A fragments are double-buffered so
// L2 latency hides under the WMMA issue stream.
// ---------------------------------------------------------------------------
__global__ __launch_bounds__(256) void stft_wmma(
    const float* __restrict__ x,
    const _Float16* __restrict__ ahg, const _Float16* __restrict__ alg,
    float* __restrict__ out)
{
    // [hi/lo][frame-subtile][n*KPAD + k] ; 53,248 bytes
    __shared__ __align__(32) _Float16 ldsB[2][2][KPAD * 16];

    int b  = blockIdx.x / N32_TILES;
    int nt = blockIdx.x % N32_TILES;
    int t0 = nt * WG_FRAMES;
    const float* xb = x + (size_t)b * N_SAMP;

    // Stage B: gather strided/overlapping windows, split f32 -> f16 hi+lo.
    for (int e = threadIdx.x; e < KPAD * WG_FRAMES; e += 256) {
        int k = e % KPAD;                  // consecutive tid -> consecutive k -> coalesced
        int n = e / KPAD;
        int t = t0 + n;
        long long s = (long long)t * HOP + k - HALF_NFFT;
        float v = 0.0f;
        if (k < N_FFT && t < N_FRAMES && s >= 0 && s < N_SAMP) v = xb[s];
        _Float16 h = (_Float16)v;
        _Float16 l = (_Float16)(v - (float)h);
        int nsub = n >> 4, nn = n & 15;
        ldsB[0][nsub][nn * KPAD + k] = h;
        ldsB[1][nsub][nn * KPAD + k] = l;
    }
    __syncthreads();

    int lane = threadIdx.x & 31;
    // wave-uniform scalars (SALU control flow; EXEC stays all-1 for WMMA)
    int wvu  = __builtin_amdgcn_readfirstlane(threadIdx.x >> 5);
    int nsub = wvu & 1;                    // which 16-frame subtile
    int tsel = wvu >> 1;                   // wave owns tasks tsel, tsel+4, ..., tsel+24

    v8f acc[7] = {};

    // prime the A double-buffer with (task=tsel, chunk=0)
    size_t a0 = (((size_t)tsel * K_CHUNKS + 0) * 32 + lane) * 16;
    v16h ah = *(const v16h*)(ahg + a0);
    v16h al = *(const v16h*)(alg + a0);

    for (int c = 0; c < K_CHUNKS; ++c) {
        const int brow = (lane & 15) * KPAD + c * 32 + (lane >> 4) * 16;
        v16h bh = *(const v16h*)&ldsB[0][nsub][brow];
        v16h bl = *(const v16h*)&ldsB[1][nsub][brow];
#pragma unroll
        for (int i = 0; i < 7; ++i) {
            // prefetch next task's fragments (or next chunk's first task)
            v16h ah_n = ah, al_n = al;
            int nc = (i == 6) ? c + 1 : c;
            int ntask = (i == 6) ? tsel : tsel + (i + 1) * 4;
            if (nc < K_CHUNKS) {
                size_t aoff = (((size_t)ntask * K_CHUNKS + nc) * 32 + lane) * 16;
                ah_n = *(const v16h*)(ahg + aoff);
                al_n = *(const v16h*)(alg + aoff);
            }
            acc[i] = __builtin_amdgcn_wmma_f32_16x16x32_f16(
                        false, ah, false, bh, (short)0, acc[i], false, false);
            acc[i] = __builtin_amdgcn_wmma_f32_16x16x32_f16(
                        false, ah, false, bl, (short)0, acc[i], false, false);
            acc[i] = __builtin_amdgcn_wmma_f32_16x16x32_f16(
                        false, al, false, bh, (short)0, acc[i], false, false);
            ah = ah_n; al = al_n;
        }
    }

    // Store C fragments: lanes 0-15 -> rows m..m+7, lanes 16-31 -> rows m+8..m+15,
    // column = lane&15. Consecutive lanes hit consecutive floats (64B runs).
    int nn    = lane & 15;
    int t     = t0 + nsub * 16 + nn;
    int rbase = (lane >> 4) * 8;
#pragma unroll
    for (int i = 0; i < 7; ++i) {
        int task = tsel + i * 4;
        if (task < N_TASKS && t < N_FRAMES) {
            int mat = task / M_TILES;      // 0 = real(cos), 1 = imag(sin)
            int mt  = task % M_TILES;
            float* o = out + (size_t)mat * ((size_t)N_BATCH * N_FREQ * N_FRAMES)
                           + (size_t)b * ((size_t)N_FREQ * N_FRAMES) + t;
#pragma unroll
            for (int r = 0; r < 8; ++r) {
                int M = mt * 16 + rbase + r;
                if (M < N_FREQ) o[(size_t)M * N_FRAMES] = acc[i][r];
            }
        }
    }
}

extern "C" void kernel_launch(void* const* d_in, const int* in_sizes, int n_in,
                              void* d_out, int out_size, void* d_ws, size_t ws_size,
                              hipStream_t stream) {
    const float* x  = (const float*)d_in[0];   // [32,1,480000]
    const float* ck = (const float*)d_in[1];   // [201,1,400]
    const float* sk = (const float*)d_in[2];   // [201,1,400]
    float* out = (float*)d_out;                // real[32,201,3001] ++ imag[32,201,3001]

    _Float16* ahg = (_Float16*)d_ws;           // 186368 halves (A hi plane, 28 padded tasks)
    _Float16* alg = ahg + A_ELEMS;             // 186368 halves (A lo plane)

    stft_pack_kernels<<<(A_ELEMS + 255) / 256, 256, 0, stream>>>(ck, sk, ahg, alg);
    stft_wmma<<<N_BATCH * N32_TILES, 256, 0, stream>>>(x, ahg, alg, out);
}
